// MambaBlock_74646531604854
// MI455X (gfx1250) — matrix operand order repaired and checked by reading, not compiled, not run on hardware
//
#include <hip/hip_runtime.h>
#include <hip/hip_bf16.h>

// ---------------------------------------------------------------------------
// MI455X (gfx1250, wave32) implementation of the hybrid attention+Mamba block.
// All large GEMMs and the attention core use v_wmma_f32_16x16x32_f16.
// ---------------------------------------------------------------------------

typedef __attribute__((ext_vector_type(16))) _Float16 v16h;
typedef __attribute__((ext_vector_type(8)))  _Float16 v8h;
typedef __attribute__((ext_vector_type(4)))  _Float16 v4h;
typedef __attribute__((ext_vector_type(8)))  float    v8f;

#define WMMA_F16(a, b, c) \
  __builtin_amdgcn_wmma_f32_16x16x32_f16(false, (a), false, (b), (short)0, (c), false, false)

__device__ __forceinline__ v16h mkfrag8(const _Float16* lo, const _Float16* hi) {
  union { v16h v; v8h h[2]; } u;
  u.h[0] = *(const v8h*)lo;
  u.h[1] = *(const v8h*)hi;
  return u.v;
}

__device__ __forceinline__ float sigmoidf_(float x) { return 1.f / (1.f + __expf(-x)); }
__device__ __forceinline__ float siluf_(float x)    { return x * sigmoidf_(x); }

// 16-lane row max reduction on the VALU via DPP (keeps the two half-wave row
// groups independent, exactly matching the WMMA C-matrix layout). Falls back
// to ds_bpermute-based shuffles if the builtin is unavailable.
#if __has_builtin(__builtin_amdgcn_update_dpp)
template <int CTRL>
__device__ __forceinline__ float dpp_mov_(float x) {
  union { float f; int i; } a, r;
  a.f = x;
  r.i = __builtin_amdgcn_update_dpp(a.i, a.i, CTRL, 0xF, 0xF, false);
  return r.f;
}
__device__ __forceinline__ float rowmax16_(float x) {
  x = fmaxf(x, dpp_mov_<0xB1>(x));    // quad_perm [1,0,3,2]  (xor 1)
  x = fmaxf(x, dpp_mov_<0x4E>(x));    // quad_perm [2,3,0,1]  (xor 2)
  x = fmaxf(x, dpp_mov_<0x141>(x));   // row_half_mirror      (xor 7)
  x = fmaxf(x, dpp_mov_<0x140>(x));   // row_mirror           (xor 15)
  return x;
}
#else
__device__ __forceinline__ float rowmax16_(float x) {
  for (int d = 1; d < 16; d <<= 1) x = fmaxf(x, __shfl_xor(x, d, 32));
  return x;
}
#endif

// ---------------------------------------------------------------------------
// Tiled WMMA GEMM:  C[4096,N] = act(A[4096,K] @ Bw[K,N] + bias)
// Block tile 256(M) x 64(N), K-step 32, 256 threads = 8 waves.
// Wave computes 32x64 -> 2 A-frags x 4 B-frags = 8 WMMA per K-step.
// GATHER: A logical row m -> (m & ~1023) + rowIdx[m & 1023]  (seq permutation).
// ACT: 0=none, 1=silu, 2=softplus. No bounds checks: requires N%64==0, K%32==0.
// ---------------------------------------------------------------------------
template <bool GATHER, bool BIAS, int ACT>
__global__ __launch_bounds__(256)
void gemm_wmma_f16(const float* __restrict__ A, const float* __restrict__ Bw,
                   const float* __restrict__ bias, float* __restrict__ C,
                   int K, int lda, int ldb, int ldc, const int* __restrict__ rowIdx)
{
  __shared__ __align__(64) _Float16 As[256][32]; // row-major [m][k]   (16 KB)
  __shared__ __align__(64) _Float16 Bs[64][32];  // K-major  [n][k]    (4 KB)

  const int tid  = threadIdx.x;
  const int wave = tid >> 5;
  const int lane = tid & 31;
  const int hf   = lane >> 4;
  const int lr   = lane & 15;

  const int mBase = blockIdx.y * 256;
  const int nBase = blockIdx.x * 64;

  v8f acc0[4], acc1[4];
#pragma unroll
  for (int j = 0; j < 4; ++j)
#pragma unroll
    for (int i = 0; i < 8; ++i) { acc0[j][i] = 0.f; acc1[j][i] = 0.f; }

  for (int k0 = 0; k0 < K; k0 += 32) {
    __syncthreads();
    // A tile: 256 rows x 32 cols as float4 groups (2048 groups, 8/thread)
#pragma unroll
    for (int i = 0; i < 8; ++i) {
      int e  = tid + i * 256;
      int r  = e >> 3;            // 8 groups of 4 per row
      int c4 = (e & 7) * 4;
      int m  = mBase + r;
      if (GATHER) m = (m & ~1023) + rowIdx[m & 1023];
      const float4 v = *(const float4*)(A + (size_t)m * lda + k0 + c4);
      v4h hv = { (_Float16)v.x, (_Float16)v.y, (_Float16)v.z, (_Float16)v.w };
      *(v4h*)&As[r][c4] = hv;
    }
    // B tile: 32 x 64 as float4 groups (512 groups, 2/thread), store K-major
#pragma unroll
    for (int i = 0; i < 2; ++i) {
      int e  = tid + i * 256;
      int k  = e >> 4;            // 16 groups of 4 per k-row
      int n4 = (e & 15) * 4;
      const float4 v = *(const float4*)(Bw + (size_t)(k0 + k) * ldb + nBase + n4);
      Bs[n4 + 0][k] = (_Float16)v.x;
      Bs[n4 + 1][k] = (_Float16)v.y;
      Bs[n4 + 2][k] = (_Float16)v.z;
      Bs[n4 + 3][k] = (_Float16)v.w;
    }
    // prefetch next K-tile (speculative; dropped if OOB)
    {
      int m = mBase + tid;
      if (GATHER) m = (m & ~1023) + rowIdx[m & 1023];
      __builtin_prefetch(A + (size_t)m * lda + k0 + 32, 0, 1);
      if (tid < 32) __builtin_prefetch(Bw + (size_t)(k0 + 32 + tid) * ldb + nBase, 0, 1);
    }
    __syncthreads();

    // A frags: lane holds row, K {hf*8..+7} and {16+hf*8..+7}
    const _Float16* ar0 = &As[wave * 32 + lr][0];
    const _Float16* ar1 = &As[wave * 32 + 16 + lr][0];
    v16h af0 = mkfrag8(ar0 + hf * 8, ar0 + 16 + hf * 8);
    v16h af1 = mkfrag8(ar1 + hf * 8, ar1 + 16 + hf * 8);
#pragma unroll
    for (int j = 0; j < 4; ++j) {
      // B frag: lane holds column (j*16+lr), K {hf*8..+7} and {16+hf*8..+7}
      v16h bf = mkfrag8(&Bs[j * 16 + lr][hf * 8], &Bs[j * 16 + lr][16 + hf * 8]);
      acc0[j] = WMMA_F16(af0, bf, acc0[j]);
      acc1[j] = WMMA_F16(af1, bf, acc1[j]);
    }
  }

  // Epilogue: C-layout -> VGPR r holds row r (lanes<16) / r+8 (lanes>=16)
#pragma unroll
  for (int g = 0; g < 2; ++g) {
#pragma unroll
    for (int j = 0; j < 4; ++j) {
#pragma unroll
      for (int r = 0; r < 8; ++r) {
        int row = mBase + wave * 32 + g * 16 + r + hf * 8;
        int col = nBase + j * 16 + lr;
        float v = g ? acc1[j][r] : acc0[j][r];
        if (BIAS) v += bias[col];
        if (ACT == 1) v = siluf_(v);
        if (ACT == 2) v = (v > 15.f) ? v : log1pf(__expf(v));
        C[(size_t)row * ldc + col] = v;
      }
    }
  }
}

// ---------------------------------------------------------------------------
// Flash attention (no mask): B=4, H=8, L=1024, DH=64.
// Block = 256 thr = 8 waves; wave owns 16 queries x 64 features.
// Streams keys/values in 32-key chunks through LDS, online softmax.
// Row sums computed by a 5th "ones-column" V tile (extra WMMA) instead of
// cross-lane shuffles; row max via DPP (VALU) -> softmax uses no LDS pipe.
// 9 WMMA per chunk per wave. qkv layout: (B, L, 3*512).
// ---------------------------------------------------------------------------
__global__ __launch_bounds__(256)
void attn_wmma(const float* __restrict__ qkv, float* __restrict__ out)
{
  __shared__ __align__(64) _Float16 Kl[32][64];     // [key][feat]
  __shared__ __align__(64) _Float16 Vt[80][32];     // [feat][key]; rows 64..79 = ones column
  __shared__ __align__(64) _Float16 Ps[8][16][32];  // per-wave P scratch

  const int tid  = threadIdx.x;
  const int wave = tid >> 5;
  const int lane = tid & 31;
  const int hf   = lane >> 4;
  const int lr   = lane & 15;

  const int qt = blockIdx.x & 7;
  const int bh = blockIdx.x >> 3;
  const int b  = bh >> 3;
  const int h  = bh & 7;
  const int qBase = qt * 128 + wave * 16;
  const int QL = 3 * 512;
  const float scale = 0.125f;           // 1/sqrt(64)

  // ones-column tile: feature row 64 is 1.0, rows 65..79 are 0 (written once)
#pragma unroll
  for (int i = 0; i < 2; ++i) {
    int e = tid + i * 256;              // 512 entries
    int fr = 64 + (e >> 5), kk = e & 31;
    Vt[fr][kk] = (_Float16)((fr == 64) ? 1.f : 0.f);
  }

  // Q fragments (features 0..31 and 32..63), pre-scaled, vector loads
  v16h qf0, qf1;
  {
    const float* qp = qkv + ((size_t)(b * 1024 + qBase + lr)) * QL + h * 64;
    float4 a0 = *(const float4*)(qp + hf * 8);
    float4 a1 = *(const float4*)(qp + hf * 8 + 4);
    float4 a2 = *(const float4*)(qp + 16 + hf * 8);
    float4 a3 = *(const float4*)(qp + 16 + hf * 8 + 4);
    float4 a4 = *(const float4*)(qp + 32 + hf * 8);
    float4 a5 = *(const float4*)(qp + 32 + hf * 8 + 4);
    float4 a6 = *(const float4*)(qp + 48 + hf * 8);
    float4 a7 = *(const float4*)(qp + 48 + hf * 8 + 4);
    union { v16h v; _Float16 e[16]; } u0, u1;
    u0.e[0]=(_Float16)(a0.x*scale); u0.e[1]=(_Float16)(a0.y*scale);
    u0.e[2]=(_Float16)(a0.z*scale); u0.e[3]=(_Float16)(a0.w*scale);
    u0.e[4]=(_Float16)(a1.x*scale); u0.e[5]=(_Float16)(a1.y*scale);
    u0.e[6]=(_Float16)(a1.z*scale); u0.e[7]=(_Float16)(a1.w*scale);
    u0.e[8]=(_Float16)(a2.x*scale); u0.e[9]=(_Float16)(a2.y*scale);
    u0.e[10]=(_Float16)(a2.z*scale); u0.e[11]=(_Float16)(a2.w*scale);
    u0.e[12]=(_Float16)(a3.x*scale); u0.e[13]=(_Float16)(a3.y*scale);
    u0.e[14]=(_Float16)(a3.z*scale); u0.e[15]=(_Float16)(a3.w*scale);
    u1.e[0]=(_Float16)(a4.x*scale); u1.e[1]=(_Float16)(a4.y*scale);
    u1.e[2]=(_Float16)(a4.z*scale); u1.e[3]=(_Float16)(a4.w*scale);
    u1.e[4]=(_Float16)(a5.x*scale); u1.e[5]=(_Float16)(a5.y*scale);
    u1.e[6]=(_Float16)(a5.z*scale); u1.e[7]=(_Float16)(a5.w*scale);
    u1.e[8]=(_Float16)(a6.x*scale); u1.e[9]=(_Float16)(a6.y*scale);
    u1.e[10]=(_Float16)(a6.z*scale); u1.e[11]=(_Float16)(a6.w*scale);
    u1.e[12]=(_Float16)(a7.x*scale); u1.e[13]=(_Float16)(a7.y*scale);
    u1.e[14]=(_Float16)(a7.z*scale); u1.e[15]=(_Float16)(a7.w*scale);
    qf0 = u0.v; qf1 = u1.v;
  }

  float mstat[8], alph[8];
  v8f oacc[4], lacc;                    // lacc col 0 accumulates row sums
#pragma unroll
  for (int r = 0; r < 8; ++r) { mstat[r] = -1e30f; lacc[r] = 0.f; }
#pragma unroll
  for (int j = 0; j < 4; ++j)
#pragma unroll
    for (int i = 0; i < 8; ++i) oacc[j][i] = 0.f;

  for (int kb = 0; kb < 32; ++kb) {
    const int keyBase = kb * 32;
    // cooperative K/V chunk load (32 keys x 64 feats) as float4 groups
#pragma unroll
    for (int i = 0; i < 2; ++i) {
      int e  = tid + i * 256;
      int kk = e >> 4;
      int f4 = (e & 15) * 4;
      const float* kvp = qkv + ((size_t)(b * 1024 + keyBase + kk)) * QL + h * 64 + f4;
      float4 kv = *(const float4*)(kvp + 512);
      float4 vv = *(const float4*)(kvp + 1024);
      v4h kh = { (_Float16)kv.x, (_Float16)kv.y, (_Float16)kv.z, (_Float16)kv.w };
      *(v4h*)&Kl[kk][f4] = kh;
      Vt[f4 + 0][kk] = (_Float16)vv.x;
      Vt[f4 + 1][kk] = (_Float16)vv.y;
      Vt[f4 + 2][kk] = (_Float16)vv.z;
      Vt[f4 + 3][kk] = (_Float16)vv.w;
    }
    __syncthreads();

    // S = Q K^T : two 16x16 tiles (keys 0-15, 16-31), K-dim 64 = 2 frags
    v8f s0, s1;
#pragma unroll
    for (int i = 0; i < 8; ++i) { s0[i] = 0.f; s1[i] = 0.f; }
    {
      v16h bf;
      bf = mkfrag8(&Kl[lr][hf * 8], &Kl[lr][16 + hf * 8]);            s0 = WMMA_F16(qf0, bf, s0);
      bf = mkfrag8(&Kl[lr][32 + hf * 8], &Kl[lr][48 + hf * 8]);       s0 = WMMA_F16(qf1, bf, s0);
      bf = mkfrag8(&Kl[16 + lr][hf * 8], &Kl[16 + lr][16 + hf * 8]);  s1 = WMMA_F16(qf0, bf, s1);
      bf = mkfrag8(&Kl[16 + lr][32 + hf * 8], &Kl[16 + lr][48 + hf * 8]);  s1 = WMMA_F16(qf1, bf, s1);
    }

    // online softmax: row max via DPP (VALU only); sums via ones-column WMMA
#pragma unroll
    for (int r = 0; r < 8; ++r) {
      float mx = rowmax16_(fmaxf(s0[r], s1[r]));
      float mn = fmaxf(mstat[r], mx);
      alph[r]  = __expf(mstat[r] - mn);
      mstat[r] = mn;
      float p0 = __expf(s0[r] - mn);
      float p1 = __expf(s1[r] - mn);
      Ps[wave][r + 8 * hf][lr]      = (_Float16)p0;
      Ps[wave][r + 8 * hf][16 + lr] = (_Float16)p1;
    }

    // P (16x32) as A fragment; same-wave LDS ops are ordered
    v16h pf = mkfrag8(&Ps[wave][lr][hf * 16], &Ps[wave][lr][hf * 16 + 8]);
#pragma unroll
    for (int j = 0; j < 4; ++j) {
#pragma unroll
      for (int r = 0; r < 8; ++r) oacc[j][r] *= alph[r];
      v16h vf = *(const v16h*)&Vt[j * 16 + lr][hf * 16];
      oacc[j] = WMMA_F16(pf, vf, oacc[j]);
    }
    // l = l*alpha + rowsum(P) via the ones column
    {
#pragma unroll
      for (int r = 0; r < 8; ++r) lacc[r] *= alph[r];
      v16h vf = *(const v16h*)&Vt[64 + lr][hf * 16];
      lacc = WMMA_F16(pf, vf, lacc);
    }
    __syncthreads();
  }

  // broadcast row sums (col 0 lives in lane 0 / lane 16 of each half)
  float linv[8];
#pragma unroll
  for (int r = 0; r < 8; ++r) linv[r] = 1.f / __shfl(lacc[r], hf * 16, 32);

#pragma unroll
  for (int j = 0; j < 4; ++j)
#pragma unroll
    for (int r = 0; r < 8; ++r) {
      int q = qBase + r + 8 * hf;
      out[((size_t)(b * 1024 + q)) * 512 + h * 64 + j * 16 + lr] = oacc[j][r] * linv[r];
    }
}

// ---------------------------------------------------------------------------
// ada: out[b, j] = sum_k silu(c[b,k]) * W_ada[k,j] + b_ada[j]   (B=4, tiny)
// ---------------------------------------------------------------------------
__global__ __launch_bounds__(256)
void ada_kernel(const float* __restrict__ c, const float* __restrict__ W,
                const float* __restrict__ bias, float* __restrict__ out)
{
  int j = blockIdx.x * 256 + threadIdx.x;
  int b = blockIdx.y;
  if (j >= 1536) return;
  float s = bias[j];
  for (int k = 0; k < 1024; ++k) {
    float cv = c[b * 1024 + k];
    s += siluf_(cv) * W[(size_t)k * 1536 + j];
  }
  out[b * 1536 + j] = s;
}

// ---------------------------------------------------------------------------
// LN(x) over D=512 then two adaLN modulations. One block per (b,l) row.
// ---------------------------------------------------------------------------
__global__ __launch_bounds__(256)
void ln_mod_kernel(const float* __restrict__ x, const float* __restrict__ g1,
                   const float* __restrict__ b1, const float* __restrict__ ada1,
                   const float* __restrict__ ada2, float* __restrict__ xm1,
                   float* __restrict__ xm2)
{
  __shared__ float rs[256], rs2[256];
  const int row = blockIdx.x, tid = threadIdx.x;
  const int b = row >> 10;
  const float* xr = x + (size_t)row * 512;
  float s = 0.f, s2 = 0.f;
  for (int j = tid; j < 512; j += 256) { float v = xr[j]; s += v; s2 += v * v; }
  rs[tid] = s; rs2[tid] = s2; __syncthreads();
  for (int o = 128; o > 0; o >>= 1) {
    if (tid < o) { rs[tid] += rs[tid + o]; rs2[tid] += rs2[tid + o]; }
    __syncthreads();
  }
  float mean = rs[0] * (1.f / 512.f);
  float var  = rs2[0] * (1.f / 512.f) - mean * mean;
  float inv  = rsqrtf(var + 1e-5f);
  for (int j = tid; j < 512; j += 256) {
    float xn = (xr[j] - mean) * inv * g1[j] + b1[j];
    xm1[(size_t)row * 512 + j] = xn * (1.f + ada1[b * 1536 + 512 + j]) + ada1[b * 1536 + j];
    xm2[(size_t)row * 512 + j] = xn * (1.f + ada2[b * 1536 + 512 + j]) + ada2[b * 1536 + j];
  }
}

// ---------------------------------------------------------------------------
// LN over DM=1024 with inv_perm row gather on the input.
// ---------------------------------------------------------------------------
__global__ __launch_bounds__(256)
void ln2_kernel(const float* __restrict__ comb2, const int* __restrict__ inv_perm,
                const float* __restrict__ g, const float* __restrict__ bb,
                float* __restrict__ hn)
{
  __shared__ float rs[256], rs2[256];
  const int row = blockIdx.x, tid = threadIdx.x;
  const int b = row >> 10, l = row & 1023;
  const float* xr = comb2 + ((size_t)(b * 1024 + inv_perm[l])) * 1024;
  float s = 0.f, s2 = 0.f;
  for (int j = tid; j < 1024; j += 256) { float v = xr[j]; s += v; s2 += v * v; }
  rs[tid] = s; rs2[tid] = s2; __syncthreads();
  for (int o = 128; o > 0; o >>= 1) {
    if (tid < o) { rs[tid] += rs[tid + o]; rs2[tid] += rs2[tid + o]; }
    __syncthreads();
  }
  float mean = rs[0] * (1.f / 1024.f);
  float var  = rs2[0] * (1.f / 1024.f) - mean * mean;
  float inv  = rsqrtf(var + 1e-5f);
  for (int j = tid; j < 1024; j += 256)
    hn[(size_t)row * 1024 + j] = (xr[j] - mean) * inv * g[j] + bb[j];
}

// ---------------------------------------------------------------------------
// Depthwise causal conv (k=4, left pad 3) + bias + silu over xm = ui[:, :DI].
// ---------------------------------------------------------------------------
__global__ __launch_bounds__(256)
void conv_silu_kernel(const float* __restrict__ ui, const float* __restrict__ cw,
                      const float* __restrict__ cb, float* __restrict__ xconv)
{
  int idx = blockIdx.x * 256 + threadIdx.x;
  if (idx >= 4 * 1024 * 2048) return;
  int c = idx & 2047;
  int row = idx >> 11;
  int l = row & 1023, b = row >> 10;
  float s = cb[c];
#pragma unroll
  for (int k = 0; k < 4; ++k) {
    int ls = l + k - 3;
    if (ls >= 0) s += ui[((size_t)(b * 1024 + ls)) * 4096 + c] * cw[c * 4 + k];
  }
  xconv[idx] = siluf_(s);
}

// ---------------------------------------------------------------------------
// Selective-state scan. One thread per (b, channel); 16 states in registers.
// Fuses + D_par*x and * silu(z). B_t/C_t broadcast via LDS per timestep.
// ---------------------------------------------------------------------------
__global__ __launch_bounds__(128)
void ssm_scan_kernel(const float* __restrict__ dtb, const float* __restrict__ dbc,
                     const float* __restrict__ xm, const float* __restrict__ ui,
                     const float* __restrict__ A_log, const float* __restrict__ D_par,
                     float* __restrict__ ym)
{
  __shared__ float Bs[16], Cs[16];
  const int d = blockIdx.x * 128 + threadIdx.x;
  const int b = blockIdx.y;
  float Ac[16], hst[16];
#pragma unroll
  for (int s = 0; s < 16; ++s) { Ac[s] = -__expf(A_log[d * 16 + s]); hst[s] = 0.f; }
  const float Dp = D_par[d];
  for (int t = 0; t < 1024; ++t) {
    size_t row = (size_t)b * 1024 + t;
    if (threadIdx.x < 32) {
      float v = dbc[row * 64 + 32 + threadIdx.x];   // B at 32..47, C at 48..63
      if (threadIdx.x < 16) Bs[threadIdx.x] = v; else Cs[threadIdx.x - 16] = v;
    }
    __syncthreads();
    float dt = dtb[row * 2048 + d];
    float xt = xm[row * 2048 + d];
    float dx = dt * xt;
    float acc = 0.f;
#pragma unroll
    for (int s = 0; s < 16; ++s) {
      hst[s] = hst[s] * __expf(dt * Ac[s]) + dx * Bs[s];
      acc += hst[s] * Cs[s];
    }
    float z = ui[row * 4096 + 2048 + d];
    ym[row * 2048 + d] = (acc + Dp * xt) * siluf_(z);
    __syncthreads();
  }
}

// ---------------------------------------------------------------------------
// Per-row gating scalar: w = sigmoid(hn1_row . W_n2 + b_n2)
// ---------------------------------------------------------------------------
__global__ __launch_bounds__(128)
void wsig_kernel(const float* __restrict__ hn1, const float* __restrict__ Wn2,
                 const float* __restrict__ bn2, float* __restrict__ wsc)
{
  __shared__ float rs[128];
  const int row = blockIdx.x, tid = threadIdx.x;
  float s = 0.f;
  for (int j = tid; j < 512; j += 128) s += hn1[(size_t)row * 512 + j] * Wn2[j];
  rs[tid] = s; __syncthreads();
  for (int o = 64; o > 0; o >>= 1) { if (tid < o) rs[tid] += rs[tid + o]; __syncthreads(); }
  if (tid == 0) wsc[row] = sigmoidf_(rs[0] + bn2[0]);
}

// ---------------------------------------------------------------------------
// Final mix: out = x + gt1 * (w*att1 + (1-w)*att2)
// ---------------------------------------------------------------------------
__global__ __launch_bounds__(256)
void mix_kernel(const float* __restrict__ x, const float* __restrict__ comb,
                const float* __restrict__ wsc, const float* __restrict__ ada1,
                float* __restrict__ out)
{
  int idx = blockIdx.x * 256 + threadIdx.x;
  if (idx >= 4 * 1024 * 512) return;
  int row = idx >> 9, j = idx & 511, b = row >> 10;
  float w  = wsc[row];
  float a1 = comb[(size_t)row * 1024 + j];
  float a2 = comb[(size_t)row * 1024 + 512 + j];
  float gt = ada1[b * 1536 + 1024 + j];
  out[idx] = x[idx] + gt * (w * a1 + (1.f - w) * a2);
}

// ---------------------------------------------------------------------------
extern "C" void kernel_launch(void* const* d_in, const int* in_sizes, int n_in,
                              void* d_out, int out_size, void* d_ws, size_t ws_size,
                              hipStream_t stream)
{
  (void)in_sizes; (void)n_in; (void)out_size; (void)ws_size;
  const float* x      = (const float*)d_in[0];
  const float* c1     = (const float*)d_in[1];
  const float* c2     = (const float*)d_in[2];
  const float* g1     = (const float*)d_in[3];
  const float* b1     = (const float*)d_in[4];
  const float* W_ada  = (const float*)d_in[5];
  const float* b_ada  = (const float*)d_in[6];
  const float* W_qkv  = (const float*)d_in[7];
  const float* b_qkv  = (const float*)d_in[8];
  const float* W_proj = (const float*)d_in[9];
  const float* b_proj = (const float*)d_in[10];
  const float* W_in   = (const float*)d_in[11];
  const float* conv_w = (const float*)d_in[12];
  const float* conv_b = (const float*)d_in[13];
  const float* W_xprj = (const float*)d_in[14];
  const float* W_dt   = (const float*)d_in[15];
  const float* b_dt   = (const float*)d_in[16];
  const float* A_log  = (const float*)d_in[17];
  const float* D_par  = (const float*)d_in[18];
  const float* W_out  = (const float*)d_in[19];
  const float* g_nm   = (const float*)d_in[20];
  const float* b_nm   = (const float*)d_in[21];
  const float* W_n1   = (const float*)d_in[22];
  const float* b_n1   = (const float*)d_in[23];
  const float* W_n2   = (const float*)d_in[24];
  const float* b_n2   = (const float*)d_in[25];
  const int*   perm   = (const int*)d_in[26];
  const int*   invp   = (const int*)d_in[27];
  float* out = (float*)d_out;

  float* ws = (float*)d_ws;
  size_t off = 0;
  auto alloc = [&](size_t n) { float* p = ws + off; off += n; return p; };
  float* ada1  = alloc(4 * 1536);
  float* ada2  = alloc(4 * 1536);
  float* xmod1 = alloc((size_t)4096 * 512);   // later reused as attnO1
  float* xmod2 = alloc((size_t)4096 * 512);   // later reused as attnO2
  float* qkv1  = alloc((size_t)4096 * 1536);  // later reused as hn
  float* qkv2  = alloc((size_t)4096 * 1536);  // later reused as hn1
  float* combA = alloc((size_t)4096 * 1024);  // [att1 | att2]
  float* ui    = alloc((size_t)4096 * 4096);  // [xm_raw | z]; later reused as comb2
  float* xconv = alloc((size_t)4096 * 2048);
  float* dbc   = alloc((size_t)4096 * 64);
  float* dtb   = alloc((size_t)4096 * 2048);
  float* ym    = alloc((size_t)4096 * 2048);
  float* wsc   = alloc(4096);
  float* attnO1 = xmod1;
  float* attnO2 = xmod2;
  float* comb2  = ui;
  float* hn     = qkv1;
  float* hn1    = qkv2;

  // 1) adaLN parameters from conditioning vectors
  ada_kernel<<<dim3(6, 4), 256, 0, stream>>>(c1, W_ada, b_ada, ada1);
  ada_kernel<<<dim3(6, 4), 256, 0, stream>>>(c2, W_ada, b_ada, ada2);

  // 2) LN + two modulations
  ln_mod_kernel<<<4096, 256, 0, stream>>>(x, g1, b1, ada1, ada2, xmod1, xmod2);

  // 3) QKV projections (N=1536, K=512)
  gemm_wmma_f16<false, true, 0><<<dim3(1536 / 64, 16), 256, 0, stream>>>(
      xmod1, W_qkv, b_qkv, qkv1, 512, 512, 1536, 1536, nullptr);
  gemm_wmma_f16<false, true, 0><<<dim3(1536 / 64, 16), 256, 0, stream>>>(
      xmod2, W_qkv, b_qkv, qkv2, 512, 512, 1536, 1536, nullptr);

  // 4) Flash attention (B*H*(L/128) = 256 blocks)
  attn_wmma<<<256, 256, 0, stream>>>(qkv1, attnO1);
  attn_wmma<<<256, 256, 0, stream>>>(qkv2, attnO2);

  // 5) Output projections into concatenated comb buffer (ldc = 1024)
  gemm_wmma_f16<false, true, 0><<<dim3(512 / 64, 16), 256, 0, stream>>>(
      attnO1, W_proj, b_proj, combA, 512, 512, 512, 1024, nullptr);
  gemm_wmma_f16<false, true, 0><<<dim3(512 / 64, 16), 256, 0, stream>>>(
      attnO2, W_proj, b_proj, combA + 512, 512, 512, 512, 1024, nullptr);

  // 6) W_in with perm row gather (N=4096, K=1024)
  gemm_wmma_f16<true, false, 0><<<dim3(4096 / 64, 16), 256, 0, stream>>>(
      combA, W_in, nullptr, ui, 1024, 1024, 4096, 4096, perm);

  // 7) depthwise causal conv + silu
  conv_silu_kernel<<<(4 * 1024 * 2048) / 256, 256, 0, stream>>>(ui, conv_w, conv_b, xconv);

  // 8) x-projection -> [dt(32) | B(16) | C(16)]  (N=64, K=2048)
  gemm_wmma_f16<false, false, 0><<<dim3(1, 16), 256, 0, stream>>>(
      xconv, W_xprj, nullptr, dbc, 2048, 2048, 64, 64, nullptr);

  // 9) dt = softplus(dt_raw @ W_dt + b_dt)  (N=2048, K=32, lda=64)
  gemm_wmma_f16<false, true, 2><<<dim3(2048 / 64, 16), 256, 0, stream>>>(
      dbc, W_dt, b_dt, dtb, 32, 64, 2048, 2048, nullptr);

  // 10) sequential SSM scan, fused with D_par*x and silu(z) gate
  ssm_scan_kernel<<<dim3(2048 / 128, 4), 128, 0, stream>>>(
      dtb, dbc, xconv, ui, A_log, D_par, ym);

  // 11) W_out (N=1024, K=2048) into comb2 (reuses ui storage)
  gemm_wmma_f16<false, false, 0><<<dim3(1024 / 64, 16), 256, 0, stream>>>(
      ym, W_out, nullptr, comb2, 2048, 2048, 1024, 1024, nullptr);

  // 12) LN with inv_perm gather
  ln2_kernel<<<4096, 256, 0, stream>>>(comb2, invp, g_nm, b_nm, hn);

  // 13) W_n1 + silu (N=512, K=1024)
  gemm_wmma_f16<false, true, 1><<<dim3(512 / 64, 16), 256, 0, stream>>>(
      hn, W_n1, b_n1, hn1, 1024, 1024, 512, 512, nullptr);

  // 14) gating scalar per token
  wsig_kernel<<<4096, 128, 0, stream>>>(hn1, W_n2, b_n2, wsc);

  // 15) final mix + residual
  mix_kernel<<<(4 * 1024 * 512) / 256, 256, 0, stream>>>(x, combA, wsc, ada1, out);
}
